// MultiHeadAtt_20323785245304
// MI455X (gfx1250) — compile-verified, matrix operand order checked by reading, not run on hardware
//
#include <hip/hip_runtime.h>
#include <hip/hip_bf16.h>

// ---------------------------------------------------------------------------
// MHA forward for MI455X (gfx1250, wave32, WMMA).
// out = ctx @ Wo + bo, att = softmax(mask(Q@K^T/sqrt(dk)))
// Matmuls via v_wmma_f32_16x16x32_f16. Score->softmax->ctx fused through LDS
// so the 537MB att tensor is written to HBM exactly once (NT stores).
// ---------------------------------------------------------------------------

#define BATCH 4
#define TSEQ  2048
#define NF    512
#define NHEAD 8
#define DKH   64
#define SROW  2052   // LDS score row stride (floats): 4-float pad -> b128 reads
                     // from 16 different rows cover all 64 LDS banks.

typedef __attribute__((ext_vector_type(4)))  float    v4f;
typedef __attribute__((ext_vector_type(8)))  float    v8f;
typedef __attribute__((ext_vector_type(4)))  _Float16 v4h;
typedef __attribute__((ext_vector_type(8)))  _Float16 v8h;
typedef __attribute__((ext_vector_type(16))) _Float16 v16h;

// Fragment builders following CDNA5 ISA 7.12.2 16-bit 16x32 layouts:
// lane<16 holds K = base+{0..7, 16..23}; lane>=16 holds K = base+{8..15, 24..31}.
// Caller passes p already offset by (lane<16 ? 0 : 8) in K.
__device__ __forceinline__ v16h frag_f16(const _Float16* p) {
  v8h lo = *(const v8h*)(p);
  v8h hi = *(const v8h*)(p + 16);
  v16h r;
#pragma unroll
  for (int i = 0; i < 8; ++i) { r[i] = lo[i]; r[8 + i] = hi[i]; }
  return r;
}

__device__ __forceinline__ v16h frag_f32(const float* p) {
  v4f a0 = *(const v4f*)(p);
  v4f a1 = *(const v4f*)(p + 4);
  v4f a2 = *(const v4f*)(p + 16);
  v4f a3 = *(const v4f*)(p + 20);
  v16h r;
#pragma unroll
  for (int i = 0; i < 4; ++i) {
    r[i]      = (_Float16)a0[i];
    r[4 + i]  = (_Float16)a1[i];
    r[8 + i]  = (_Float16)a2[i];
    r[12 + i] = (_Float16)a3[i];
  }
  return r;
}

__device__ __forceinline__ v8f wmma16(v16h a, v16h b, v8f c) {
  return __builtin_amdgcn_wmma_f32_16x16x32_f16(false, a, false, b, (short)0, c,
                                                false, false);
}

// ---------------------------------------------------------------------------
// 1) Weight transpose + f16 convert: Wt[n][k] = (f16) W[k][n]   (512x512)
// ---------------------------------------------------------------------------
__global__ __launch_bounds__(256) void wt_kernel(const float* __restrict__ W,
                                                 _Float16* __restrict__ Wt) {
  int idx = blockIdx.x * 256 + threadIdx.x;   // idx = n*512 + k
  int n = idx >> 9;
  int k = idx & 511;
  Wt[idx] = (_Float16)W[k * NF + n];
}

// ---------------------------------------------------------------------------
// 2) Projection: P = X[8192x512] @ W + b  -> f16, head layouts.
//    mode 0: Qh[b][h][t][d]   mode 1: Kh[b][h][t][d] (*1/sqrt(dk))
//    mode 2: Vht[b][h][d][t]  (transposed: att@V B-fragments become contiguous)
// Wave = one 16x64 strip (4 N-tiles reuse one A fragment). Grid: (8, 64).
// ---------------------------------------------------------------------------
__global__ __launch_bounds__(256) void proj_kernel(
    const float* __restrict__ X, const _Float16* __restrict__ Wt,
    const float* __restrict__ bias, _Float16* __restrict__ Out, int mode,
    float scale) {
  const int wave = threadIdx.x >> 5;
  const int lane = threadIdx.x & 31;
  const int lane16 = lane & 15;
  const int kof = (lane < 16) ? 0 : 8;
  const int tileM = blockIdx.y * 128 + wave * 16;
  const int tileN = blockIdx.x * 64;

  v8f acc[4] = {};
  const float*    arow = X  + (size_t)(tileM + lane16) * NF + kof;
  const _Float16* brow = Wt + (size_t)(tileN + lane16) * NF + kof;

  for (int kb = 0; kb < NF; kb += 32) {
    v16h af = frag_f32(arow + kb);
#pragma unroll
    for (int t = 0; t < 4; ++t) {
      v16h bf = frag_f16(brow + (size_t)t * 16 * NF + kb);
      acc[t] = wmma16(af, bf, acc[t]);
    }
  }

  const int mbase = tileM + ((lane < 16) ? 0 : 8);
#pragma unroll
  for (int t = 0; t < 4; ++t) {
    int col = tileN + t * 16 + lane16;
    float bv = bias[col];
    int h = col >> 6, d = col & 63;
#pragma unroll
    for (int j = 0; j < 8; ++j) {
      int row = mbase + j;                 // global row in [0, B*T)
      int b = row >> 11, tt = row & (TSEQ - 1);
      float val = (acc[t][j] + bv) * scale;
      size_t oidx;
      if (mode == 2)
        oidx = (((size_t)b * NHEAD + h) * DKH + d) * TSEQ + tt;
      else
        oidx = (((size_t)b * NHEAD + h) * TSEQ + tt) * DKH + d;
      Out[oidx] = (_Float16)val;
    }
  }
}

// ---------------------------------------------------------------------------
// 3) Fused scores -> softmax -> att(NT store, written once) -> ctx.
// Block = 256 threads (8 waves), owns (b,h) and 16 query rows.
//  P1: each wave computes 16x256 of the score strip (WMMA) -> LDS (f32,
//      padded stride), mask -> -inf.
//  P2: wave w softmaxes rows 2w,2w+1 in LDS; streams final att row to HBM.
//  P3: ctx = scores(16x2048, LDS) @ Vht(2048x64): K split across waves,
//      partials reduced via LDS (score buffer reused after a barrier).
// Grid: (T/16 = 128, B*H = 32). Dynamic LDS: 16*SROW*4 = 131328 B.
// ---------------------------------------------------------------------------
__global__ __launch_bounds__(256) void fused_att_kernel(
    const _Float16* __restrict__ Qh, const _Float16* __restrict__ Kh,
    const _Float16* __restrict__ Vht, const int* __restrict__ mask,
    float* __restrict__ att, _Float16* __restrict__ ctx16) {
  extern __shared__ float smem[];
  const int bh = blockIdx.y;
  const int b = bh >> 3, h = bh & 7;
  const int tileM = blockIdx.x * 16;
  const int tid = threadIdx.x;
  const int wave = tid >> 5, lane = tid & 31, lane16 = lane & 15;
  const int kof = (lane < 16) ? 0 : 8;
  const int rbase = (lane < 16) ? 0 : 8;

  const _Float16* Qb = Qh  + (size_t)bh * TSEQ * DKH;
  const _Float16* Kb = Kh  + (size_t)bh * TSEQ * DKH;
  const _Float16* Vb = Vht + (size_t)bh * DKH * TSEQ;

  // ---- Phase 1: scores for cols [wave*256, wave*256+256) ----
  const _Float16* qrow = Qb + (size_t)(tileM + lane16) * DKH + kof;
  v16h af0 = frag_f16(qrow);
  v16h af1 = frag_f16(qrow + 32);

  for (int ct = 0; ct < 4; ++ct) {
    int tileN = wave * 256 + ct * 64;
    v8f acc[4] = {};
#pragma unroll
    for (int t = 0; t < 4; ++t) {
      const _Float16* bp = Kb + (size_t)(tileN + t * 16 + lane16) * DKH + kof;
      acc[t] = wmma16(af0, frag_f16(bp), acc[t]);
      acc[t] = wmma16(af1, frag_f16(bp + 32), acc[t]);
    }
#pragma unroll
    for (int t = 0; t < 4; ++t) {
      int col = tileN + t * 16 + lane16;
      bool msk = (mask[b * TSEQ + col] == 0);
#pragma unroll
      for (int j = 0; j < 8; ++j) {
        float v2 = msk ? -__builtin_inff() : acc[t][j];
        smem[(rbase + j) * SROW + col] = v2;
      }
    }
  }
  __syncthreads();

  // ---- Phase 2: softmax rows 2*wave, 2*wave+1; stream att to HBM once ----
  for (int rr = 0; rr < 2; ++rr) {
    int row = wave * 2 + rr;
    float* prow = smem + row * SROW;
    float mx = -__builtin_inff();
    for (int i = lane; i < TSEQ; i += 32) mx = fmaxf(mx, prow[i]);
#pragma unroll
    for (int off = 16; off > 0; off >>= 1)
      mx = fmaxf(mx, __shfl_xor(mx, off, 32));
    float s = 0.f;
    for (int i = lane; i < TSEQ; i += 32) s += expf(prow[i] - mx);
#pragma unroll
    for (int off = 16; off > 0; off >>= 1) s += __shfl_xor(s, off, 32);
    float inv = 1.0f / s;
    float* gout = att + ((size_t)bh * TSEQ + tileM + row) * TSEQ;
    for (int i = lane; i < TSEQ; i += 32) {
      float o = expf(prow[i] - mx) * inv;
      if (mask[b * TSEQ + i] == 0) o = 0.f;
      prow[i] = o;
      __builtin_nontemporal_store(o, gout + i);
    }
  }
  __syncthreads();

  // ---- Phase 3: ctx partials, K range [wave*256, wave*256+256) ----
  v8f cacc[4] = {};
  const int kwb = wave * 256;
#pragma unroll
  for (int kk = 0; kk < 8; ++kk) {
    int kb = kwb + kk * 32;
    v16h af = frag_f32(smem + lane16 * SROW + kb + kof);  // ds_load_b128
#pragma unroll
    for (int t = 0; t < 4; ++t) {
      const _Float16* bp = Vb + (size_t)(t * 16 + lane16) * TSEQ + kb + kof;
      cacc[t] = wmma16(af, frag_f16(bp), cacc[t]);
    }
  }
  __syncthreads();  // all waves done reading scores -> safe to reuse LDS

  // partials: P[wave][row*64 + d]
#pragma unroll
  for (int t = 0; t < 4; ++t)
#pragma unroll
    for (int j = 0; j < 8; ++j)
      smem[wave * 1024 + (rbase + j) * 64 + t * 16 + lane16] = cacc[t][j];
  __syncthreads();

  // reduce 8 partials -> 16x64 ctx tile; 4 outputs per thread
  v4f s4 = {};
#pragma unroll
  for (int w2 = 0; w2 < 8; ++w2)
    s4 += *(const v4f*)(smem + w2 * 1024 + tid * 4);
  int oid = tid * 4;
  int row = oid >> 6, d0 = oid & 63;
  v4h h4;
#pragma unroll
  for (int i = 0; i < 4; ++i) h4[i] = (_Float16)s4[i];
  *(v4h*)(ctx16 + ((size_t)b * TSEQ + tileM + row) * NF + h * DKH + d0) = h4;
}

// ---------------------------------------------------------------------------
// 4) out = ctx16[8192x512] @ Wo + bo -> f32. Grid: (8, 64).
// ---------------------------------------------------------------------------
__global__ __launch_bounds__(256) void outproj_kernel(
    const _Float16* __restrict__ ctx16, const _Float16* __restrict__ Wot,
    const float* __restrict__ bias, float* __restrict__ out) {
  const int wave = threadIdx.x >> 5;
  const int lane = threadIdx.x & 31;
  const int lane16 = lane & 15;
  const int kof = (lane < 16) ? 0 : 8;
  const int tileM = blockIdx.y * 128 + wave * 16;
  const int tileN = blockIdx.x * 64;

  v8f acc[4] = {};
  const _Float16* arow = ctx16 + (size_t)(tileM + lane16) * NF + kof;
  const _Float16* brow = Wot   + (size_t)(tileN + lane16) * NF + kof;

  for (int kb = 0; kb < NF; kb += 32) {
    v16h af = frag_f16(arow + kb);
#pragma unroll
    for (int t = 0; t < 4; ++t) {
      v16h bf = frag_f16(brow + (size_t)t * 16 * NF + kb);
      acc[t] = wmma16(af, bf, acc[t]);
    }
  }

  const int mbase = tileM + ((lane < 16) ? 0 : 8);
#pragma unroll
  for (int t = 0; t < 4; ++t) {
    int col = tileN + t * 16 + lane16;
    float bv = bias[col];
#pragma unroll
    for (int j = 0; j < 8; ++j) {
      int row = mbase + j;
      out[(size_t)row * NF + col] = acc[t][j] + bv;
    }
  }
}

// ---------------------------------------------------------------------------
extern "C" void kernel_launch(void* const* d_in, const int* in_sizes, int n_in,
                              void* d_out, int out_size, void* d_ws,
                              size_t ws_size, hipStream_t stream) {
  const float* q    = (const float*)d_in[0];
  const float* k    = (const float*)d_in[1];
  const float* v    = (const float*)d_in[2];
  const int*   mask = (const int*)d_in[3];
  const float* Wq = (const float*)d_in[4];
  const float* bq = (const float*)d_in[5];
  const float* Wk = (const float*)d_in[6];
  const float* bk = (const float*)d_in[7];
  const float* Wv = (const float*)d_in[8];
  const float* bv = (const float*)d_in[9];
  const float* Wo = (const float*)d_in[10];
  const float* bo = (const float*)d_in[11];

  float* out = (float*)d_out;                                  // [B,T,NF]
  float* att = out + (size_t)BATCH * TSEQ * NF;                // [B,H,T,T]

  const size_t WT  = (size_t)NF * NF;
  const size_t BHT = (size_t)BATCH * NHEAD * TSEQ * DKH;
  _Float16* Wqt   = (_Float16*)d_ws;
  _Float16* Wkt   = Wqt + WT;
  _Float16* Wvt   = Wkt + WT;
  _Float16* Wot   = Wvt + WT;
  _Float16* Qh    = Wot + WT;
  _Float16* Kh    = Qh + BHT;
  _Float16* Vht   = Kh + BHT;
  _Float16* ctx16 = Vht + BHT;   // total ~34 MiB

  const float kscale = 0.125f;   // 1/sqrt(64)
  const int   shmem  = 16 * SROW * 4;  // 131328 B dynamic LDS (<=320KB/WGP)
  // opt-in for >64KB dynamic LDS (not a stream op; graph-capture safe)
  (void)hipFuncSetAttribute((const void*)fused_att_kernel,
                            hipFuncAttributeMaxDynamicSharedMemorySize, shmem);

  // 1) weight transpose + f16
  wt_kernel<<<(NF * NF) / 256, 256, 0, stream>>>(Wq, Wqt);
  wt_kernel<<<(NF * NF) / 256, 256, 0, stream>>>(Wk, Wkt);
  wt_kernel<<<(NF * NF) / 256, 256, 0, stream>>>(Wv, Wvt);
  wt_kernel<<<(NF * NF) / 256, 256, 0, stream>>>(Wo, Wot);

  // 2) projections
  dim3 gproj(NF / 64, (BATCH * TSEQ) / 128);
  proj_kernel<<<gproj, 256, 0, stream>>>(q, Wqt, bq, Qh, 0, 1.0f);
  proj_kernel<<<gproj, 256, 0, stream>>>(k, Wkt, bk, Kh, 1, kscale);
  proj_kernel<<<gproj, 256, 0, stream>>>(v, Wvt, bv, Vht, 2, 1.0f);

  // 3) fused scores + softmax + att store + ctx
  dim3 gfuse(TSEQ / 16, BATCH * NHEAD);
  fused_att_kernel<<<gfuse, 256, shmem, stream>>>(Qh, Kh, Vht, mask, att,
                                                  ctx16);

  // 4) output projection
  outproj_kernel<<<gproj, 256, 0, stream>>>(ctx16, Wot, bo, out);
}
